// WindowAttention3d3_36481452212837
// MI455X (gfx1250) — matrix operand order repaired and checked by reading, not compiled, not run on hardware
//
#include <hip/hip_runtime.h>

// ---------------------------------------------------------------------------
// WindowAttention (Swin-ish with rolled neighbor keys) for MI455X / gfx1250
//   K1: qkv = x @ qkv_w^T + qkv_b            (f32 in, f16 out, WMMA f16)
//   K2: per (window, head) attention          (WMMA f16, wave32 softmax,
//                                              async global->LDS gathers)
//   K3: out = attn @ proj_w^T + proj_b        (f16 in, f32 out, WMMA f16,
//                                              async A-tile staging)
// ---------------------------------------------------------------------------

typedef _Float16 h8  __attribute__((ext_vector_type(8)));
typedef _Float16 h16 __attribute__((ext_vector_type(16)));
typedef float    f8  __attribute__((ext_vector_type(8)));
typedef float    f4  __attribute__((ext_vector_type(4)));
typedef unsigned int u32x4 __attribute__((ext_vector_type(4)));

#define BDIM   8
#define HWDIM  56
#define CDIM   512
#define NHEAD  16
#define HD     32
#define WS7    7
#define WA     49      // 7*7 tokens per window
#define NROLL  132     // valid rolled neighbor keys
#define LKEY   181     // WA + NROLL
#define LPAD   192     // LKEY padded to 16-multiple (12 N tiles, 6 K steps)
#define NTOK   25088   // 8*56*56
#define C3     1536

// ---- constexpr valid-rolled-index table (matches numpy mask order) --------
struct VTab { unsigned char quad[NROLL], pi[NROLL], pj[NROLL]; };
constexpr VTab make_vtab() {
  VTab t{};
  int c = 0;
  for (int q = 0; q < 4; ++q)
    for (int i = 0; i < 7; ++i)
      for (int j = 0; j < 7; ++j) {
        bool zero = false;
        if (q == 0) zero = (i < 4) && (j < 4);   // m_tl[:-3,:-3]=0
        if (q == 1) zero = (i < 4) && (j >= 3);  // m_tr[:-3, 3:]=0
        if (q == 2) zero = (i >= 3) && (j < 4);  // m_bl[ 3:,:-3]=0
        if (q == 3) zero = (i >= 3) && (j >= 3); // m_br[ 3:, 3:]=0
        if (!zero) { t.quad[c] = (unsigned char)q; t.pi[c] = (unsigned char)i;
                     t.pj[c] = (unsigned char)j; ++c; }
      }
  return t;
}
__constant__ VTab VALID = make_vtab();

// ---- CDNA5 async global->LDS copy (GV mode, tracked by ASYNCcnt) ----------
// VDST = per-lane LDS byte address; VADDR = per-lane 64-bit global address.
// Low 32 bits of a generic pointer to a __shared__ object are the LDS offset
// (flat-aperture rule), so a truncation is enough.
__device__ __forceinline__ unsigned lds_addr(const void* p) {
  return (unsigned)(unsigned long long)p;
}
__device__ __forceinline__ void async_load_b128(unsigned lds, const void* g) {
  asm volatile("global_load_async_to_lds_b128 %0, %1, off"
               :: "v"(lds), "v"(g) : "memory");
}
__device__ __forceinline__ void wait_async0() {
  asm volatile("s_wait_asynccnt 0" ::: "memory");
}

// ---- WMMA helpers ---------------------------------------------------------
__device__ __forceinline__ f8 fzero8() {
  f8 z;
#pragma unroll
  for (int i = 0; i < 8; ++i) z[i] = 0.0f;
  return z;
}

__device__ __forceinline__ f8 wmma16(h16 a, h16 b, f8 c) {
  // D = A(16x32 f16) * B(32x16 f16) + C(16x16 f32)
  return __builtin_amdgcn_wmma_f32_16x16x32_f16(false, a, false, b,
                                                (short)0, c, false, false);
}

// A fragment: 16x32 f16, row-major source with K contiguous (stride in halfs).
// lane&15 = row, lane>>4 selects K half: lo = K[h*8 .. h*8+7], hi = K[16+h*8 ..]
__device__ __forceinline__ h16 load_frag_a(const _Float16* base, int stride, int lane) {
  const int r = lane & 15, h = lane >> 4;
  const _Float16* p = base + r * stride + h * 8;
  h8 lo = *(const h8*)(p);
  h8 hi = *(const h8*)(p + 16);
  return __builtin_shufflevector(lo, hi, 0,1,2,3,4,5,6,7,8,9,10,11,12,13,14,15);
}

// B fragment: 32x16 f16; source stored column-major-in-N (Bt[n][k], K contiguous).
// lane&15 = column n, lane>>4 selects K half: lo = Bt[n][h*8 .. h*8+7], hi = Bt[n][16+h*8 ..]
__device__ __forceinline__ h16 load_frag_b(const _Float16* base, int stride, int lane) {
  const int nn = lane & 15, h = lane >> 4;
  const _Float16* p = base + nn * stride + h * 8;
  h8 lo = *(const h8*)(p);
  h8 hi = *(const h8*)(p + 16);
  return __builtin_shufflevector(lo, hi, 0,1,2,3,4,5,6,7,8,9,10,11,12,13,14,15);
}

// load 16 consecutive f32, convert to f16, store 32B to LDS
__device__ __forceinline__ void cvt_store16(_Float16* dst, const float* src) {
  f4 a = *(const f4*)(src);
  f4 b = *(const f4*)(src + 4);
  f4 c = *(const f4*)(src + 8);
  f4 d = *(const f4*)(src + 12);
  h8 lo, hi;
#pragma unroll
  for (int i = 0; i < 4; ++i) {
    lo[i] = (_Float16)a[i]; lo[4 + i] = (_Float16)b[i];
    hi[i] = (_Float16)c[i]; hi[4 + i] = (_Float16)d[i];
  }
  *(h8*)(dst) = lo;
  *(h8*)(dst + 8) = hi;
}

// ---------------------------------------------------------------------------
// Kernel 1: qkv = x @ qkv_w^T + qkv_b ;  f16 output (NTOK x 1536)
// grid (196, 12), block 256 (8 waves). 128x128 tile, K step 32.
// f32->f16 conversion must pass through VALU, so tiles are staged with
// vector loads + cvt + ds stores; next K-tile is prefetched (global_prefetch).
// ---------------------------------------------------------------------------
__global__ __launch_bounds__(256) void qkv_gemm_kernel(
    const float* __restrict__ X, const float* __restrict__ Wq,
    const float* __restrict__ bq, _Float16* __restrict__ O) {
  __shared__ _Float16 As[128 * 32];
  __shared__ _Float16 Bs[128 * 32];
  const int tid = threadIdx.x;
  const int bm = blockIdx.x, bn = blockIdx.y;
  const int wid = tid >> 5, lane = tid & 31;
  const int wm = wid >> 2, wn = wid & 3;        // wave tile: 64 rows x 32 cols
  const int nn = lane & 15, hh = lane >> 4;

  f8 acc[4][2];
#pragma unroll
  for (int a = 0; a < 4; ++a) { acc[a][0] = fzero8(); acc[a][1] = fzero8(); }

  const int lrow = tid >> 1, lq = tid & 1;      // 2 threads per LDS row
  const float* ga = X  + (size_t)(bm * 128 + lrow) * CDIM + lq * 16;
  const float* gb = Wq + (size_t)(bn * 128 + lrow) * CDIM + lq * 16;
  _Float16* sa = As + lrow * 32 + lq * 16;
  _Float16* sb = Bs + lrow * 32 + lq * 16;

  for (int kk = 0; kk < 16; ++kk) {
    __syncthreads();
    cvt_store16(sa, ga);
    cvt_store16(sb, gb);
    __builtin_prefetch(ga + 32, 0, 3);          // next K tile -> global_prefetch
    __builtin_prefetch(gb + 32, 0, 3);
    ga += 32; gb += 32;
    __syncthreads();
    h16 bf0 = load_frag_b(Bs + (wn * 32 + 0)  * 32, 32, lane);
    h16 bf1 = load_frag_b(Bs + (wn * 32 + 16) * 32, 32, lane);
#pragma unroll
    for (int mt = 0; mt < 4; ++mt) {
      h16 af = load_frag_a(As + (wm * 64 + mt * 16) * 32, 32, lane);
      acc[mt][0] = wmma16(af, bf0, acc[mt][0]);
      acc[mt][1] = wmma16(af, bf1, acc[mt][1]);
    }
  }

#pragma unroll
  for (int mt = 0; mt < 4; ++mt)
#pragma unroll
    for (int nt = 0; nt < 2; ++nt) {
      const int col = bn * 128 + wn * 32 + nt * 16 + nn;
      const float bias = bq[col];
#pragma unroll
      for (int i = 0; i < 8; ++i) {
        const int row = bm * 128 + wm * 64 + mt * 16 + i + 8 * hh;
        O[(size_t)row * C3 + col] = (_Float16)(acc[mt][nt][i] + bias);
      }
    }
}

// ---------------------------------------------------------------------------
// Kernel 2: attention per (window, head). grid (16, 512), block 128 (4 waves).
// Q/K gathers are per-lane-addressed pure copies -> async global->LDS (GV).
// ---------------------------------------------------------------------------
__global__ __launch_bounds__(128) void attn_kernel(
    const _Float16* __restrict__ qkvF, const float* __restrict__ rpb_table,
    const float* __restrict__ rpb_nb, _Float16* __restrict__ aout) {
  __shared__ _Float16 Qs[64 * 32];      // queries, zero-padded rows 49..63
  __shared__ _Float16 Ks[LPAD * 32];    // keys row-major (K contiguous per key)
  __shared__ _Float16 Vt[32 * LPAD];    // V transposed (key contiguous per d)
  __shared__ _Float16 Pt[64 * LPAD];    // softmax probs f16

  const int tid  = threadIdx.x;
  const int head = blockIdx.x;
  const int win  = blockIdx.y;
  const int bimg = win >> 6;
  const int wi   = win & 63;
  const int wh0  = (wi >> 3) * WS7;
  const int ww0  = (wi & 7) * WS7;

  // ---- zero pad regions ----
  for (int t = tid; t < (64 - WA) * 32; t += 128) Qs[WA * 32 + t] = (_Float16)0.0f;
  for (int t = tid; t < (LPAD - LKEY) * 32; t += 128) Ks[LKEY * 32 + t] = (_Float16)0.0f;
  for (int t = tid; t < 32 * (LPAD - LKEY); t += 128) {
    const int d = t / (LPAD - LKEY), j = t % (LPAD - LKEY);
    Vt[d * LPAD + LKEY + j] = (_Float16)0.0f;
  }

  // ---- gather Q (49 rows x 32 halfs; 4 x 16B async chunks per row) ----
  for (int t = tid; t < WA * 4; t += 128) {
    const int j = t >> 2, c = t & 3;
    const int h = wh0 + j / 7, w = ww0 + j % 7;
    const size_t m = (size_t)((bimg * HWDIM + h) * HWDIM + w);
    async_load_b128(lds_addr(Qs + j * 32 + c * 8),
                    qkvF + m * C3 + head * HD + c * 8);
  }

  // ---- gather K (async, row-major) and V (sync load + transpose store) ----
  for (int t = tid; t < LKEY * 4; t += 128) {
    const int j = t >> 2, c = t & 3;
    int h, w;
    if (j < WA) {
      h = wh0 + j / 7; w = ww0 + j % 7;
    } else {
      const int u  = j - WA;
      const int qd = VALID.quad[u];
      const int pi = VALID.pi[u], pj = VALID.pj[u];
      const int s0 = (qd >= 2) ? 3 : -3;       // shifts: (-3,-3)(-3,3)(3,-3)(3,3)
      const int s1 = (qd & 1) ? 3 : -3;
      h = (wh0 + pi - s0 + HWDIM) % HWDIM;     // jnp.roll source coordinate
      w = (ww0 + pj - s1 + HWDIM) % HWDIM;
    }
    const size_t m = (size_t)((bimg * HWDIM + h) * HWDIM + w);
    const _Float16* kv = qkvF + m * C3 + head * HD + c * 8;
    async_load_b128(lds_addr(Ks + j * 32 + c * 8), kv + CDIM);   // k channels
    h8 vv = *(const h8*)(kv + 2 * CDIM);                         // v channels
#pragma unroll
    for (int d = 0; d < 8; ++d) Vt[(c * 8 + d) * LPAD + j] = vv[d];
  }
  wait_async0();          // ASYNCcnt -> 0: all async LDS writes landed
  __syncthreads();

  const int wid = tid >> 5, lane = tid & 31;
  const int nn = lane & 15, hh = lane >> 4;

  // ---- S = Q K^T : one WMMA per N tile (head dim 32 == one K step) ----
  h16 qa = load_frag_a(Qs + wid * 16 * 32, 32, lane);
  f8 s[12];
#pragma unroll
  for (int nt = 0; nt < 12; ++nt) {
    h16 kb = load_frag_b(Ks + nt * 16 * 32, 32, lane);
    s[nt] = wmma16(qa, kb, fzero8());
  }

  // ---- bias + masked softmax (row = wid*16 + i + 8*hh, col = nt*16 + nn) ----
  const float SC = 0.17677669529663687f;   // 1/sqrt(32)
  float rmax[8], rsum[8];
#pragma unroll
  for (int i = 0; i < 8; ++i) rmax[i] = -3.0e38f;
#pragma unroll
  for (int nt = 0; nt < 12; ++nt) {
    const int j = nt * 16 + nn;
#pragma unroll
    for (int i = 0; i < 8; ++i) {
      const int q = wid * 16 + i + 8 * hh;
      float val = -3.0e38f;
      if (q < WA && j < LKEY) {
        float bias;
        if (j < WA) {
          const int qi = q / 7, qj = q - qi * 7;
          const int ki = j / 7, kj = j - ki * 7;
          const int ridx = (qi - ki + 6) * 13 + (qj - kj + 6);
          bias = rpb_table[ridx * NHEAD + head];
        } else {
          bias = rpb_nb[((size_t)head * WA + q) * NROLL + (j - WA)];
        }
        val = s[nt][i] * SC + bias;
      }
      s[nt][i] = val;
      rmax[i] = fmaxf(rmax[i], val);
    }
  }
#pragma unroll
  for (int i = 0; i < 8; ++i) {
#pragma unroll
    for (int d = 1; d < 16; d <<= 1)
      rmax[i] = fmaxf(rmax[i], __shfl_xor(rmax[i], d, 32));
    rsum[i] = 0.0f;
  }
#pragma unroll
  for (int nt = 0; nt < 12; ++nt) {
#pragma unroll
    for (int i = 0; i < 8; ++i) {
      const float p = __expf(s[nt][i] - rmax[i]);
      rsum[i] += p;
      Pt[(wid * 16 + i + 8 * hh) * LPAD + nt * 16 + nn] = (_Float16)p;
    }
  }
#pragma unroll
  for (int i = 0; i < 8; ++i) {
#pragma unroll
    for (int d = 1; d < 16; d <<= 1) rsum[i] += __shfl_xor(rsum[i], d, 32);
  }
  __syncthreads();

  // ---- O = P V : 6 K steps x 2 N tiles ----
  f8 o0 = fzero8(), o1 = fzero8();
#pragma unroll
  for (int ks = 0; ks < 6; ++ks) {
    h16 pa = load_frag_a(Pt + wid * 16 * LPAD + ks * 32, LPAD, lane);
    h16 v0 = load_frag_b(Vt + 0  * LPAD + ks * 32, LPAD, lane);
    h16 v1 = load_frag_b(Vt + 16 * LPAD + ks * 32, LPAD, lane);
    o0 = wmma16(pa, v0, o0);
    o1 = wmma16(pa, v1, o1);
  }

  // ---- normalize (row mapping identical to S fragment -> rsum in regs) ----
#pragma unroll
  for (int i = 0; i < 8; ++i) {
    const int q = wid * 16 + i + 8 * hh;
    if (q < WA) {
      const float inv = 1.0f / rsum[i];
      _Float16* op = aout + ((size_t)win * WA + q) * CDIM + head * HD;
      op[nn]      = (_Float16)(o0[i] * inv);
      op[16 + nn] = (_Float16)(o1[i] * inv);
    }
  }
}

// ---------------------------------------------------------------------------
// Kernel 3: out = attn @ proj_w^T + proj_b ; f16 in, f32 out (NTOK x 512)
// grid (196, 4), block 256. A tile is f16 already -> async global->LDS copy.
// ---------------------------------------------------------------------------
__global__ __launch_bounds__(256) void proj_gemm_kernel(
    const _Float16* __restrict__ A, const float* __restrict__ Wp,
    const float* __restrict__ bp, float* __restrict__ O) {
  __shared__ _Float16 As[128 * 32];
  __shared__ _Float16 Bs[128 * 32];
  const int tid = threadIdx.x;
  const int bm = blockIdx.x, bn = blockIdx.y;
  const int wid = tid >> 5, lane = tid & 31;
  const int wm = wid >> 2, wn = wid & 3;
  const int nn = lane & 15, hh = lane >> 4;

  f8 acc[4][2];
#pragma unroll
  for (int a = 0; a < 4; ++a) { acc[a][0] = fzero8(); acc[a][1] = fzero8(); }

  const int lrow = tid >> 1, lq = tid & 1;
  const _Float16* ga = A  + (size_t)(bm * 128 + lrow) * CDIM + lq * 16;
  const float*    gb = Wp + (size_t)(bn * 128 + lrow) * CDIM + lq * 16;
  _Float16* sa = As + lrow * 32 + lq * 16;
  _Float16* sb = Bs + lrow * 32 + lq * 16;

  for (int kk = 0; kk < 16; ++kk) {
    __syncthreads();
    async_load_b128(lds_addr(sa),     ga);      // f16 tile: async copy, no VGPRs
    async_load_b128(lds_addr(sa) + 16, ga + 8);
    cvt_store16(sb, gb);                        // f32 weights: convert via VALU
    __builtin_prefetch(gb + 32, 0, 3);
    ga += 32; gb += 32;
    wait_async0();
    __syncthreads();
    h16 bf0 = load_frag_b(Bs + (wn * 32 + 0)  * 32, 32, lane);
    h16 bf1 = load_frag_b(Bs + (wn * 32 + 16) * 32, 32, lane);
#pragma unroll
    for (int mt = 0; mt < 4; ++mt) {
      h16 af = load_frag_a(As + (wm * 64 + mt * 16) * 32, 32, lane);
      acc[mt][0] = wmma16(af, bf0, acc[mt][0]);
      acc[mt][1] = wmma16(af, bf1, acc[mt][1]);
    }
  }

#pragma unroll
  for (int mt = 0; mt < 4; ++mt)
#pragma unroll
    for (int nt = 0; nt < 2; ++nt) {
      const int col = bn * 128 + wn * 32 + nt * 16 + nn;
      const float bias = bp[col];
#pragma unroll
      for (int i = 0; i < 8; ++i) {
        const int row = bm * 128 + wm * 64 + mt * 16 + i + 8 * hh;
        O[(size_t)row * CDIM + col] = acc[mt][nt][i] + bias;
      }
    }
}

// ---------------------------------------------------------------------------
extern "C" void kernel_launch(void* const* d_in, const int* in_sizes, int n_in,
                              void* d_out, int out_size, void* d_ws, size_t ws_size,
                              hipStream_t stream) {
  const float* x         = (const float*)d_in[0];  // (8,56,56,512)
  const float* qkv_w     = (const float*)d_in[1];  // (1536,512)
  const float* qkv_b     = (const float*)d_in[2];  // (1536,)
  const float* proj_w    = (const float*)d_in[3];  // (512,512)
  const float* proj_b    = (const float*)d_in[4];  // (512,)
  const float* rpb_table = (const float*)d_in[5];  // (169,16)
  const float* rpb_nb    = (const float*)d_in[6];  // (1,16,49,132)

  // workspace: qkv f16 (77.1 MB) + attn-out f16 (25.7 MB) ~= 103 MB
  _Float16* qkvF = (_Float16*)d_ws;
  _Float16* aout = (_Float16*)((char*)d_ws + (size_t)NTOK * C3 * sizeof(_Float16));
  float* out = (float*)d_out;                      // (512,49,512) window order

  qkv_gemm_kernel<<<dim3(NTOK / 128, C3 / 128), 256, 0, stream>>>(x, qkv_w, qkv_b, qkvF);
  attn_kernel<<<dim3(NHEAD, 512), 128, 0, stream>>>(qkvF, rpb_table, rpb_nb, aout);
  proj_gemm_kernel<<<dim3(NTOK / 128, CDIM / 128), 256, 0, stream>>>(aout, proj_w, proj_b, out);
}
